// DynamicReflectionAttention_53772990546060
// MI455X (gfx1250) — matrix operand order repaired and checked by reading, not compiled, not run on hardware
//
#include <hip/hip_runtime.h>
#include <hip/hip_bf16.h>
#include <math.h>

typedef __attribute__((ext_vector_type(16))) __bf16 bfx16;
typedef __attribute__((ext_vector_type(8)))  float  fx8;
typedef __attribute__((ext_vector_type(2)))  float  fx2;

#define NP   32768      // total pixels (B*H*W)
#define NHW  16384      // H*W
#define ND   192        // channels

// ---- optional CDNA5 async global->LDS path (guarded; sync fallback keeps compile safe) ----
#if defined(__has_builtin)
#if __has_builtin(__builtin_amdgcn_global_load_async_to_lds_b128) && __has_builtin(__builtin_amdgcn_s_wait_asynccnt)
#define ASYNC_LDS 1
#endif
#endif
#ifndef ASYNC_LDS
#define ASYNC_LDS 0
#endif

#if ASYNC_LDS
typedef __attribute__((__vector_size__(16))) int i32x4;
typedef __attribute__((address_space(1))) i32x4 g_i32x4;   // global i32x4
typedef __attribute__((address_space(3))) i32x4 l_i32x4;   // LDS i32x4
#endif

// ---------- bf16 helpers (storage = unsigned short) ----------
__device__ __forceinline__ unsigned short f2bf(float f){
  unsigned u = __builtin_bit_cast(unsigned, f);
  unsigned r = u + 0x7FFFu + ((u >> 16) & 1u);      // round-to-nearest-even
  return (unsigned short)(r >> 16);
}
__device__ __forceinline__ float bf2f(unsigned short h){
  unsigned u = ((unsigned)h) << 16;
  return __builtin_bit_cast(float, u);
}

// ---------- WMMA wrappers ----------
__device__ __forceinline__ fx8 wmma_bf(bfx16 a, bfx16 b, fx8 c){
  return __builtin_amdgcn_wmma_f32_16x16x32_bf16(false, a, false, b, (short)0, c, false, false);
}
__device__ __forceinline__ fx8 wmma_f32(fx2 a, fx2 b, fx8 c){
  return __builtin_amdgcn_wmma_f32_16x16x4_f32(false, a, false, b, (short)0, c, false, false);
}

// ---------- bf16 fragment loaders (contiguous-k layouts: k = k0 + 16*half + e, wave32) ----------
// A 16x32 (MxK): lane m=L%16, half=L/16; elems -> K=k0+half*16+e
__device__ __forceinline__ bfx16 load_a(const unsigned short* A, int ld, int m0, int k0, int lane){
  int m = m0 + (lane & 15);
  int half = lane >> 4;
  const unsigned short* p = A + (size_t)m * ld + k0 + half * 16;
  union { bfx16 v; uint4 q[2]; } u;
  u.q[0] = *(const uint4*)(p);
  u.q[1] = *(const uint4*)(p + 8);
  return u.v;
}
// B 32x16 (KxN) from global row-major W (OxK): lane n=L%16, half=L/16; elems -> K=k0+half*16+e
__device__ __forceinline__ bfx16 load_b(const unsigned short* W, int ld, int n0, int k0, int lane){
  int n = n0 + (lane & 15);
  int half = lane >> 4;
  const unsigned short* p = W + (size_t)n * ld + k0 + half * 16;
  union { bfx16 v; uint4 q[2]; } u;
  u.q[0] = *(const uint4*)(p);
  u.q[1] = *(const uint4*)(p + 8);
  return u.v;
}
// B fragment from LDS tile: 64 rows, SLD bf16 stride (pow2 bytes), XOR-16B swizzle per row
template<int SLD>
__device__ __forceinline__ bfx16 load_b_lds(const unsigned short* sW, int n0, int k0, int lane){
  int n = n0 + (lane & 15);
  int half = lane >> 4;
  int swz = (lane & 15) * 16;                 // bytes
  const char* row = (const char*)(sW + (size_t)n * SLD);
  int cb = (k0 + half * 16) * 2;              // byte offset in row
  union { bfx16 v; uint4 q[2]; } u;
  u.q[0] = *(const uint4*)(row + ( cb        ^ swz));
  u.q[1] = *(const uint4*)(row + ((cb + 16) ^ swz));
  return u.v;
}

// stage a 64 x K bf16 weight tile (rows n0..n0+63 of W) into LDS with swizzle
template<int K, int SLD, int CH>
__device__ __forceinline__ void stage_w(unsigned short* sW, const unsigned short* W, int n0){
  const char* g = (const char*)(W + (size_t)n0 * K);
  char* l = (char*)sW;
  #pragma unroll
  for (int i = 0; i < CH; ++i){
    int c  = i * 256 + threadIdx.x;           // 16B chunk id
    int n  = c / (K / 8);
    int cb = (c % (K / 8)) * 16;
    int lo = n * (SLD * 2) + (cb ^ ((n & 15) * 16));
    int go = n * (K * 2) + cb;
#if ASYNC_LDS
    __builtin_amdgcn_global_load_async_to_lds_b128((g_i32x4*)(g + go), (l_i32x4*)(l + lo), 0, 0);
#else
    *(uint4*)(l + lo) = *(const uint4*)(g + go);
#endif
  }
#if ASYNC_LDS
  __builtin_amdgcn_s_wait_asynccnt(0);
#endif
  __syncthreads();
}

// wave computes a 16(M) x 64(N) strip: one A fragment feeds 4 WMMAs
template<int K, int SLD>
__device__ __forceinline__ void mm_wave64(const unsigned short* A, const unsigned short* sW,
                                          int m0, int lane, fx8 acc[4]){
  #pragma unroll
  for (int k = 0; k < K; k += 32){
    bfx16 a = load_a(A, K, m0, k, lane);
    #pragma unroll
    for (int j = 0; j < 4; ++j){
      bfx16 b = load_b_lds<SLD>(sW, j * 16, k, lane);
      acc[j] = wmma_bf(a, b, acc[j]);
    }
  }
}

// ---------- kernels ----------
__global__ void k_f2bf(const float* __restrict__ s, unsigned short* __restrict__ d, int n){
  int i = blockIdx.x * 256 + threadIdx.x;
  if (i < n) d[i] = f2bf(s[i]);
}

// channel-major fp32 input (b,c,hw) -> per-pixel layernorm -> pixel-major bf16 [p][192]
__global__ void k_cnorm_in(const float* __restrict__ F0, const float* __restrict__ F1,
                           unsigned short* __restrict__ D0, unsigned short* __restrict__ D1){
  int p = blockIdx.x * 256 + threadIdx.x;
  const float* F = blockIdx.y ? F1 : F0;
  unsigned short* D = blockIdx.y ? D1 : D0;
  int b = p >> 14, hw = p & (NHW - 1);
  const float* src = F + (size_t)b * ND * NHW + hw;
  float s = 0.f, s2 = 0.f;
  for (int c = 0; c < ND; ++c){ float v = src[(size_t)c * NHW]; s += v; s2 += v * v; }
  float mu = s * (1.f / ND);
  float inv = rsqrtf(s2 * (1.f / ND) - mu * mu + 1e-5f);
  unsigned short* dst = D + (size_t)p * ND;
  for (int c = 0; c < ND; ++c){ float v = src[(size_t)c * NHW]; dst[c] = f2bf((v - mu) * inv); }
}

// pixel-major fp32 [p][192] -> layernorm -> pixel-major bf16 (wave per pixel)
__global__ void k_cnorm_pm(const float* __restrict__ src, unsigned short* __restrict__ dst){
  int lane = threadIdx.x & 31;
  int p = blockIdx.x * 8 + (threadIdx.x >> 5);
  const float* row = src + (size_t)p * ND;
  float v[6], s = 0.f, s2 = 0.f;
  #pragma unroll
  for (int i = 0; i < 6; ++i){ v[i] = row[lane + i * 32]; s += v[i]; s2 += v[i] * v[i]; }
  #pragma unroll
  for (int m = 16; m > 0; m >>= 1){ s += __shfl_xor(s, m, 32); s2 += __shfl_xor(s2, m, 32); }
  float mu = s * (1.f / ND);
  float inv = rsqrtf(s2 * (1.f / ND) - mu * mu + 1e-5f);
  unsigned short* d = dst + (size_t)p * ND;
  #pragma unroll
  for (int i = 0; i < 6; ++i) d[lane + i * 32] = f2bf((v[i] - mu) * inv);
}

// GEMM (M=NP,K=192) -> channel-major fp32 out[oc][p]; block = 128M x 64N, 8 waves
__global__ void __launch_bounds__(256) k_gemm_cm64(const unsigned short* __restrict__ A,
                                                   const unsigned short* __restrict__ W,
                                                   float* __restrict__ Out){
  __shared__ unsigned short sW[64 * 256];
  stage_w<192, 256, 6>(sW, W, blockIdx.y * 64);
  int wave = threadIdx.x >> 5, lane = threadIdx.x & 31, half = lane >> 4;
  int m0 = blockIdx.x * 128 + wave * 16;
  fx8 acc[4] = {};
  mm_wave64<192, 256>(A, sW, m0, lane, acc);
  #pragma unroll
  for (int j = 0; j < 4; ++j){
    int n = blockIdx.y * 64 + j * 16 + (lane & 15);
    float* o = Out + (size_t)n * NP + m0 + half * 8;
    union { fx8 v; float4 q[2]; } u; u.v = acc[j];
    *(float4*)(o) = u.q[0];
    *(float4*)(o + 4) = u.q[1];
  }
}

// out = ctx @ w_fc^T + b_fc + FrN + FyN -> pixel-major fp32
__global__ void __launch_bounds__(256) k_gemm_fc64(const unsigned short* __restrict__ Ctx,
                                                   const unsigned short* __restrict__ Wfc,
                                                   const float* __restrict__ bfc,
                                                   const unsigned short* __restrict__ FrN,
                                                   const unsigned short* __restrict__ FyN,
                                                   float* __restrict__ Outf){
  __shared__ unsigned short sW[64 * 256];
  stage_w<192, 256, 6>(sW, Wfc, blockIdx.y * 64);
  int wave = threadIdx.x >> 5, lane = threadIdx.x & 31, half = lane >> 4;
  int m0 = blockIdx.x * 128 + wave * 16;
  fx8 acc[4] = {};
  mm_wave64<192, 256>(Ctx, sW, m0, lane, acc);
  #pragma unroll
  for (int j = 0; j < 4; ++j){
    int n = blockIdx.y * 64 + j * 16 + (lane & 15);
    float bias = bfc[n];
    #pragma unroll
    for (int r = 0; r < 8; ++r){
      int p = m0 + half * 8 + r;
      size_t idx = (size_t)p * ND + n;
      Outf[idx] = acc[j][r] + bias + bf2f(FrN[idx]) + bf2f(FyN[idx]);
    }
  }
}

// x = cnorm(out) @ w_pin^T -> channel-major bf16 X[oc][p]
__global__ void __launch_bounds__(256) k_gemm_pin64(const unsigned short* __restrict__ A,
                                                    const unsigned short* __restrict__ W,
                                                    unsigned short* __restrict__ X){
  __shared__ unsigned short sW[64 * 256];
  stage_w<192, 256, 6>(sW, W, blockIdx.y * 64);
  int wave = threadIdx.x >> 5, lane = threadIdx.x & 31, half = lane >> 4;
  int m0 = blockIdx.x * 128 + wave * 16;
  fx8 acc[4] = {};
  mm_wave64<192, 256>(A, sW, m0, lane, acc);
  #pragma unroll
  for (int j = 0; j < 4; ++j){
    int n = blockIdx.y * 64 + j * 16 + (lane & 15);
    union { unsigned short h8[8]; uint4 q; } u;
    #pragma unroll
    for (int r = 0; r < 8; ++r) u.h8[r] = f2bf(acc[j][r]);
    *(uint4*)(X + (size_t)n * NP + m0 + half * 8) = u.q;
  }
}

// final = out + Y @ w_pout^T (K=512), stored channel-major (b,c,h,w) fp32 to d_out
__global__ void __launch_bounds__(256) k_gemm_pout64(const unsigned short* __restrict__ Yp,
                                                     const unsigned short* __restrict__ W,
                                                     const float* __restrict__ Outf,
                                                     float* __restrict__ Fin){
  __shared__ unsigned short sW[64 * 512];
  stage_w<512, 512, 16>(sW, W, blockIdx.y * 64);
  int wave = threadIdx.x >> 5, lane = threadIdx.x & 31, half = lane >> 4;
  int m0 = blockIdx.x * 128 + wave * 16;
  fx8 acc[4] = {};
  mm_wave64<512, 512>(Yp, sW, m0, lane, acc);
  #pragma unroll
  for (int j = 0; j < 4; ++j){
    int n = blockIdx.y * 64 + j * 16 + (lane & 15);
    #pragma unroll
    for (int r = 0; r < 8; ++r){
      int p = m0 + half * 8 + r;
      int b = p >> 14, hw = p & (NHW - 1);
      Fin[((size_t)(b * ND + n)) * NHW + hw] = Outf[(size_t)p * ND + n] + acc[j][r];
    }
  }
}

// fused gate: alpha = sigmoid(w_g2 . gelu(W_g1 [Fr;Fy] + b_g1) + b_g2), per pixel
__global__ void __launch_bounds__(256) k_alpha(const unsigned short* __restrict__ FrN,
                                               const unsigned short* __restrict__ FyN,
                                               const unsigned short* __restrict__ Wg1,
                                               const float* __restrict__ bg1,
                                               const float* __restrict__ wg2,
                                               const float* __restrict__ bg2,
                                               float* __restrict__ alpha){
  __shared__ float smem[16];
  int lane = threadIdx.x & 31, wave = threadIdx.x >> 5, half = lane >> 4;
  int m0 = blockIdx.x * 16;
  if (threadIdx.x < 16) smem[threadIdx.x] = bg2[0];
  __syncthreads();
  float partial[8];
  #pragma unroll
  for (int r = 0; r < 8; ++r) partial[r] = 0.f;
  for (int tt = 0; tt < 6; ++tt){
    int n0 = (wave * 6 + tt) * 16;
    fx8 acc = {};
    #pragma unroll
    for (int k = 0; k < 384; k += 32){
      bfx16 a = (k < 192) ? load_a(FrN, 192, m0, k, lane) : load_a(FyN, 192, m0, k - 192, lane);
      bfx16 b = load_b(Wg1, 384, n0, k, lane);
      acc = wmma_bf(a, b, acc);
    }
    int n = n0 + (lane & 15);
    float bias = bg1[n], w2 = wg2[n];
    #pragma unroll
    for (int r = 0; r < 8; ++r){
      float x = acc[r] + bias;
      float g = 0.5f * x * (1.f + erff(x * 0.70710678118f));   // exact GELU
      partial[r] += g * w2;
    }
  }
  #pragma unroll
  for (int r = 0; r < 8; ++r) atomicAdd(&smem[half * 8 + r], partial[r]);
  __syncthreads();
  if (threadIdx.x < 16)
    alpha[m0 + threadIdx.x] = 1.f / (1.f + expf(-smem[threadIdx.x]));
}

// inverse L2 norms over HW for q/k rows: inv[tsel*384 + b*192 + c]
__global__ void k_rownorm(const float* __restrict__ rq, const float* __restrict__ rk,
                          const float* __restrict__ gq, const float* __restrict__ gk,
                          float* __restrict__ inv){
  __shared__ float red[256];
  int row = blockIdx.x;                       // 0..1535
  int tsel = row / 384, rem = row % 384, b = rem / 192, c = rem % 192;
  const float* base = (tsel == 0) ? rq : (tsel == 1) ? rk : (tsel == 2) ? gq : gk;
  const float* src = base + (size_t)c * NP + b * NHW;
  float s = 0.f;
  for (int i = threadIdx.x; i < NHW; i += 256){ float v = src[i]; s += v * v; }
  red[threadIdx.x] = s; __syncthreads();
  for (int o = 128; o > 0; o >>= 1){ if (threadIdx.x < o) red[threadIdx.x] += red[threadIdx.x + o]; __syncthreads(); }
  if (threadIdx.x == 0) inv[row] = 1.f / fmaxf(sqrtf(red[0]), 1e-12f);
}

// grams S[which][b][h][48][48] = Q . K^T over HW, fp32 WMMA 16x16x4 (one wave per 16x16 tile)
__global__ void k_gram(const float* __restrict__ rq, const float* __restrict__ rk,
                       const float* __restrict__ gq, const float* __restrict__ gk,
                       float* __restrict__ S){
  int id = blockIdx.x;                        // 144 = 2*2*4*9
  int tile = id % 9, h = (id / 9) % 4, b = (id / 36) % 2, which = id / 72;
  int i0 = (tile % 3) * 16, j0 = (tile / 3) * 16;
  const float* Q = which ? rq : gq;           // which=0: rgb_q . raw_k ; which=1: raw_q . rgb_k
  const float* Kp = which ? gk : rk;
  int lane = threadIdx.x, half = lane >> 4;
  const float* qrow = Q + (size_t)(h * 48 + i0 + (lane & 15)) * NP + b * NHW + half * 2;
  const float* krow = Kp + (size_t)(h * 48 + j0 + (lane & 15)) * NP + b * NHW + half * 2;
  fx8 acc = {};
  for (int k = 0; k < NHW; k += 4){
    fx2 a;  a.x = qrow[k];  a.y = qrow[k + 1];
    fx2 bb; bb.x = krow[k]; bb.y = krow[k + 1];
    acc = wmma_f32(a, bb, acc);
  }
  float* dst = S + (size_t)(((which * 2 + b) * 4 + h) * 2304);
  int j = j0 + (lane & 15);
  #pragma unroll
  for (int r = 0; r < 8; ++r) dst[(i0 + half * 8 + r) * 48 + j] = acc[r];
}

// scale by inv-norms & temperature, row softmax over j
__global__ void k_softmax(const float* __restrict__ S, const float* __restrict__ inv,
                          const float* __restrict__ temp, float* __restrict__ attn){
  int id = blockIdx.x;                        // 16
  int which = id >> 3, b = (id >> 2) & 1, h = id & 3;
  int i = threadIdx.x;
  if (i >= 48) return;
  size_t mb = (size_t)(((which * 2 + b) * 4 + h) * 2304);
  int qsel = which ? 0 : 2;                   // raw_q : rgb_q
  int ksel = which ? 3 : 1;                   // rgb_k : raw_k
  float iq = inv[qsel * 384 + b * 192 + h * 48 + i];
  float T = temp[h];
  float v[48], mx = -1e30f;
  for (int j = 0; j < 48; ++j){
    float ik = inv[ksel * 384 + b * 192 + h * 48 + j];
    v[j] = S[mb + i * 48 + j] * iq * ik * T;
    mx = fmaxf(mx, v[j]);
  }
  float sum = 0.f;
  for (int j = 0; j < 48; ++j){ v[j] = expf(v[j] - mx); sum += v[j]; }
  float r = 1.f / sum;
  for (int j = 0; j < 48; ++j) attn[mb + i * 48 + j] = v[j] * r;
}

// context = alpha*attn_raw@raw_v + (1-alpha)*attn_rgb@rgb_v -> pixel-major bf16 [p][192]
__global__ void __launch_bounds__(256) k_ctx(const float* __restrict__ attn,
                                             const float* __restrict__ rv,
                                             const float* __restrict__ gv,
                                             const float* __restrict__ alpha,
                                             unsigned short* __restrict__ ctx){
  int w = blockIdx.x * 8 + (threadIdx.x >> 5);
  int lane = threadIdx.x & 31;
  int hwt = w & 1023, rest = w >> 10;
  int it = rest % 3, h = (rest / 3) & 3, b = rest / 12;
  int i0 = it * 16, p0 = b * NHW + hwt * 16;
  int half = lane >> 4, n = lane & 15;
  const float* a0 = attn + (size_t)(((0 * 2 + b) * 4 + h) * 2304) + (i0 + n) * 48 + half * 2;
  const float* a1 = attn + (size_t)(((1 * 2 + b) * 4 + h) * 2304) + (i0 + n) * 48 + half * 2;
  const float* v0 = rv + (size_t)(h * 48 + half * 2) * NP + p0 + n;
  const float* v1 = gv + (size_t)(h * 48 + half * 2) * NP + p0 + n;
  fx8 aR = {}, aG = {};
  #pragma unroll
  for (int k = 0; k < 48; k += 4){
    fx2 a;  a.x = a0[k];               a.y = a0[k + 1];
    fx2 bb; bb.x = v0[(size_t)k * NP]; bb.y = v0[(size_t)(k + 1) * NP];
    aR = wmma_f32(a, bb, aR);
    a.x = a1[k];                a.y = a1[k + 1];
    bb.x = v1[(size_t)k * NP];  bb.y = v1[(size_t)(k + 1) * NP];
    aG = wmma_f32(a, bb, aG);
  }
  float al = alpha[p0 + n];
  union { unsigned short h8[8]; uint4 q; } u;
  #pragma unroll
  for (int r = 0; r < 8; ++r) u.h8[r] = f2bf(al * aR[r] + (1.f - al) * aG[r]);
  *(uint4*)(ctx + (size_t)(p0 + n) * ND + h * 48 + i0 + half * 8) = u.q;
}

// pshuffle -> dw5x5(x1), dw3x3 dil2(x2) -> mish(x2)*x1 -> punshuffle -> pixel-major bf16 Y[p][512]
__global__ void k_ffn_dw(const unsigned short* __restrict__ X, const float* __restrict__ w5,
                         const float* __restrict__ wd, unsigned short* __restrict__ Y){
  int gid = blockIdx.x * 256 + threadIdx.x;          // 2*128*256*256
  int W2 = gid & 255, H2 = (gid >> 8) & 255, cp = (gid >> 16) & 127, b = gid >> 23;
  size_t bbase = (size_t)b * NHW;
  float s1 = 0.f;
  const float* wk = w5 + cp * 25;
  #pragma unroll
  for (int u = 0; u < 5; ++u){
    int Hn = H2 + u - 2; if (Hn < 0 || Hn > 255) continue;
    int hb = Hn >> 1, dy = Hn & 1;
    #pragma unroll
    for (int v = 0; v < 5; ++v){
      int Wn = W2 + v - 2; if (Wn < 0 || Wn > 255) continue;
      int wb = Wn >> 1, dx = Wn & 1;
      s1 += bf2f(X[(size_t)(cp * 4 + dy * 2 + dx) * NP + bbase + hb * 128 + wb]) * wk[u * 5 + v];
    }
  }
  float s2 = 0.f;
  const float* wk2 = wd + cp * 9;
  #pragma unroll
  for (int u = 0; u < 3; ++u){
    int Hn = H2 + 2 * u - 2; if (Hn < 0 || Hn > 255) continue;
    int hb = Hn >> 1, dy = Hn & 1;
    #pragma unroll
    for (int v = 0; v < 3; ++v){
      int Wn = W2 + 2 * v - 2; if (Wn < 0 || Wn > 255) continue;
      int wb = Wn >> 1, dx = Wn & 1;
      s2 += bf2f(X[(size_t)(512 + cp * 4 + dy * 2 + dx) * NP + bbase + hb * 128 + wb]) * wk2[u * 3 + v];
    }
  }
  float sp = (s2 > 20.f) ? s2 : log1pf(expf(s2));
  float r = s2 * tanhf(sp) * s1;                      // mish(x2) * x1
  int cout = cp * 4 + (H2 & 1) * 2 + (W2 & 1);
  int p = (int)bbase + (H2 >> 1) * 128 + (W2 >> 1);
  Y[(size_t)p * 512 + cout] = f2bf(r);
}

extern "C" void kernel_launch(void* const* d_in, const int* in_sizes, int n_in,
                              void* d_out, int out_size, void* d_ws, size_t ws_size,
                              hipStream_t stream){
  const float* F_raw    = (const float*)d_in[0];
  const float* F_Y      = (const float*)d_in[1];
  const float* temp     = (const float*)d_in[2];
  const float* w_raw_kv = (const float*)d_in[3];
  const float* w_raw_q  = (const float*)d_in[4];
  const float* w_rgb_kv = (const float*)d_in[5];
  const float* w_rgb_q  = (const float*)d_in[6];
  const float* w_g1     = (const float*)d_in[7];
  const float* b_g1     = (const float*)d_in[8];
  const float* w_g2     = (const float*)d_in[9];
  const float* b_g2     = (const float*)d_in[10];
  const float* w_fc     = (const float*)d_in[11];
  const float* b_fc     = (const float*)d_in[12];
  const float* w_pin    = (const float*)d_in[13];
  const float* w_dw5    = (const float*)d_in[14];
  const float* w_dwd    = (const float*)d_in[15];
  const float* w_pout   = (const float*)d_in[16];
  float* Fin = (float*)d_out;

  char* ws = (char*)d_ws;
  size_t off = 0;
  auto alloc = [&](size_t bytes)->char*{ char* p = ws + off; off += (bytes + 255) & ~(size_t)255; return p; };

  unsigned short* wb_rawkv = (unsigned short*)alloc(73728u * 2);
  unsigned short* wb_rawq  = (unsigned short*)alloc(36864u * 2);
  unsigned short* wb_rgbkv = (unsigned short*)alloc(73728u * 2);
  unsigned short* wb_rgbq  = (unsigned short*)alloc(36864u * 2);
  unsigned short* wb_g1    = (unsigned short*)alloc(294912u * 2);
  unsigned short* wb_fc    = (unsigned short*)alloc(36864u * 2);
  unsigned short* wb_pin   = (unsigned short*)alloc(196608u * 2);
  unsigned short* wb_pout  = (unsigned short*)alloc(98304u * 2);
  unsigned short* FrN  = (unsigned short*)alloc((size_t)NP * ND * 2);
  unsigned short* FyN  = (unsigned short*)alloc((size_t)NP * ND * 2);
  float* rawkv = (float*)alloc((size_t)384 * NP * 4);   // raw_k = rawkv, raw_v = rawkv+192*NP
  float* rawq  = (float*)alloc((size_t)192 * NP * 4);
  float* rgbkv = (float*)alloc((size_t)384 * NP * 4);
  float* rgbq  = (float*)alloc((size_t)192 * NP * 4);
  float* inv   = (float*)alloc(1536u * 4);
  float* S     = (float*)alloc(36864u * 4);
  float* attn  = (float*)alloc(36864u * 4);
  float* alpha = (float*)alloc((size_t)NP * 4);
  unsigned short* ctx  = (unsigned short*)alloc((size_t)NP * ND * 2);
  float* outf  = (float*)alloc((size_t)NP * ND * 4);
  unsigned short* outN = (unsigned short*)alloc((size_t)NP * ND * 2);
  unsigned short* Xcm  = (unsigned short*)alloc((size_t)1024 * NP * 2);
  unsigned short* Ypm  = (unsigned short*)alloc((size_t)NP * 512 * 2);

  auto cvt = [&](const float* s, unsigned short* d, int n){
    k_f2bf<<<(n + 255) / 256, 256, 0, stream>>>(s, d, n);
  };
  cvt(w_raw_kv, wb_rawkv, 73728);
  cvt(w_raw_q,  wb_rawq,  36864);
  cvt(w_rgb_kv, wb_rgbkv, 73728);
  cvt(w_rgb_q,  wb_rgbq,  36864);
  cvt(w_g1,     wb_g1,    294912);
  cvt(w_fc,     wb_fc,    36864);
  cvt(w_pin,    wb_pin,   196608);
  cvt(w_pout,   wb_pout,  98304);

  k_cnorm_in<<<dim3(128, 2), 256, 0, stream>>>(F_raw, F_Y, FrN, FyN);

  // QKV projections: blocks of 128M x 64N
  k_gemm_cm64<<<dim3(256, 6), 256, 0, stream>>>(FrN, wb_rawkv, rawkv);   // N=384
  k_gemm_cm64<<<dim3(256, 3), 256, 0, stream>>>(FrN, wb_rawq,  rawq);    // N=192
  k_gemm_cm64<<<dim3(256, 6), 256, 0, stream>>>(FyN, wb_rgbkv, rgbkv);
  k_gemm_cm64<<<dim3(256, 3), 256, 0, stream>>>(FyN, wb_rgbq,  rgbq);

  // fused gating alpha
  k_alpha<<<2048, 256, 0, stream>>>(FrN, FyN, wb_g1, b_g1, w_g2, b_g2, alpha);

  const float* raw_k = rawkv;
  const float* raw_v = rawkv + (size_t)192 * NP;
  const float* rgb_k = rgbkv;
  const float* rgb_v = rgbkv + (size_t)192 * NP;

  k_rownorm<<<1536, 256, 0, stream>>>(rawq, raw_k, rgbq, rgb_k, inv);
  k_gram<<<144, 32, 0, stream>>>(rawq, raw_k, rgbq, rgb_k, S);
  k_softmax<<<16, 64, 0, stream>>>(S, inv, temp, attn);
  k_ctx<<<3072, 256, 0, stream>>>(attn, raw_v, rgb_v, alpha, ctx);

  k_gemm_fc64<<<dim3(256, 3), 256, 0, stream>>>(ctx, wb_fc, b_fc, FrN, FyN, outf);

  // FeedForward
  k_cnorm_pm<<<4096, 256, 0, stream>>>(outf, outN);
  k_gemm_pin64<<<dim3(256, 16), 256, 0, stream>>>(outN, wb_pin, Xcm);    // N=1024
  k_ffn_dw<<<65536, 256, 0, stream>>>(Xcm, w_dw5, w_dwd, Ypm);
  k_gemm_pout64<<<dim3(256, 3), 256, 0, stream>>>(Ypm, wb_pout, outf, Fin);
}
